// MeanAggregator_60387240181927
// MI455X (gfx1250) — compile-verified, matrix operand order checked by reading, not run on hardware
//
#include <hip/hip_runtime.h>
#include <hip/hip_bf16.h>

typedef __attribute__((ext_vector_type(16))) _Float16 v16h;
typedef __attribute__((ext_vector_type(8)))  _Float16 v8h;
typedef __attribute__((ext_vector_type(8)))  float    v8f;
typedef __attribute__((ext_vector_type(4)))  float    v4f;

#define D_IN  64
#define D_OUT 128

// ---------------------------------------------------------------- zero scratch
__global__ void ma_zero_kernel(float* __restrict__ p, int n) {
    int i = blockIdx.x * blockDim.x + threadIdx.x;
    if (i < n) p[i] = 0.0f;
}

// ------------------------------------------- W [64][128] f32 -> Wt [128][64] f16
__global__ void ma_wt_kernel(const float* __restrict__ W, _Float16* __restrict__ Wt) {
    int t = blockIdx.x * blockDim.x + threadIdx.x;
    if (t >= D_IN * D_OUT) return;
    int k = t & (D_IN - 1);
    int n = t >> 6;
    Wt[n * D_IN + k] = (_Float16)W[k * D_OUT + n];
}

// ------------------------------------------------- edge scatter (L2-resident)
// thread t -> edge e = t/64, feature f = t%64. Coalesced gather of x[nbr] and
// coalesced global_atomic_add_f32 into summed[src]. counts via one lane/edge.
__global__ void ma_scatter_kernel(const float* __restrict__ x,
                                  const int*   __restrict__ ei,   // [2][E] flat
                                  float* __restrict__ summed,
                                  float* __restrict__ counts,
                                  int E) {
    unsigned t = blockIdx.x * blockDim.x + threadIdx.x;
    unsigned e = t >> 6;
    unsigned f = t & 63u;
    if (e >= (unsigned)E) return;
    int s  = ei[e];              // edge_index[0][e] : segment id
    int nb = ei[(unsigned)E + e];// edge_index[1][e] : gather index
    atomicAdd(summed + (size_t)s * D_IN + f, x[(size_t)nb * D_IN + f]);
    if (f == 0) atomicAdd(counts + s, 1.0f);
}

// ------------------------------- fused (mean + self-loop) -> WMMA GEMM -> out
// One wave32 per FULL 16-row M-tile; 8 N-tiles of 16 cols. Entirely branch-free
// past the wave-bound check: no row guards, pure store clauses in the epilogue.
__global__ void ma_gemm_wmma_kernel(const float* __restrict__ x,
                                    const float* __restrict__ summed,
                                    const float* __restrict__ counts,
                                    const _Float16* __restrict__ Wt, // [128][64]
                                    float* __restrict__ out,
                                    int n_full_tiles) {
    int gwave = (int)((blockIdx.x * blockDim.x + threadIdx.x) >> 5);
    int lane  = (int)(threadIdx.x & 31u);
    if (gwave >= n_full_tiles) return;       // wave-uniform: EXEC stays all-1s

    int lrow  = lane & 15;                   // row / col within tile
    int hsel  = lane >> 4;                   // half-wave select
    int kbase = hsel * 8;                    // K base for A/B fragment layout

    int row = gwave * 16 + lrow;             // always < N (full tiles only)

    float inv = 1.0f / (counts[row] + 1.0f); // +1 for self loop

    // A fragments: mean row, fp16, ISA layout (K = kbase+{0..7}, kbase+16+{0..7})
    v16h a[2];
#pragma unroll
    for (int c = 0; c < 2; ++c) {
        int k0 = c * 32 + kbase;
        const v4f* s4 = (const v4f*)(summed + (size_t)row * D_IN + k0);
        const v4f* x4 = (const v4f*)(x      + (size_t)row * D_IN + k0);
        v4f m[4];
        m[0] = (s4[0] + x4[0]) * inv;   // K = k0 .. k0+3
        m[1] = (s4[1] + x4[1]) * inv;   // K = k0+4 .. k0+7
        m[2] = (s4[4] + x4[4]) * inv;   // K = k0+16 .. k0+19
        m[3] = (s4[5] + x4[5]) * inv;   // K = k0+20 .. k0+23
#pragma unroll
        for (int j = 0; j < 8; ++j) {
            a[c][j]     = (_Float16)m[j >> 2][j & 3];
            a[c][8 + j] = (_Float16)m[2 + (j >> 2)][j & 3];
        }
    }

    int m0 = gwave * 16 + hsel * 8;          // output row base, half-wave
#pragma unroll
    for (int t = 0; t < 8; ++t) {            // 8 N-tiles -> 128 output cols
        int col = t * 16 + lrow;
        const _Float16* wr = Wt + (size_t)col * D_IN;
        v8f acc = {};
#pragma unroll
        for (int c = 0; c < 2; ++c) {
            int k0 = c * 32 + kbase;
            v8h lo = *(const v8h*)(wr + k0);       // K = k0 .. k0+7
            v8h hi = *(const v8h*)(wr + k0 + 16);  // K = k0+16 .. k0+23
            v16h b;
#pragma unroll
            for (int j = 0; j < 8; ++j) { b[j] = lo[j]; b[8 + j] = hi[j]; }
            acc = __builtin_amdgcn_wmma_f32_16x16x32_f16(
                      false, a[c], false, b, (short)0, acc, false, false);
        }
        // One base address + immediate offsets (row stride 512 B < 24-bit ioffset)
        float* o = out + (size_t)m0 * D_OUT + col;
#pragma unroll
        for (int r = 0; r < 8; ++r) o[r * D_OUT] = acc[r];
    }
}

// --------------------- ragged tail (< 16 rows), launched only when N % 16 != 0
__global__ void ma_tail_kernel(const float* __restrict__ x,
                               const float* __restrict__ summed,
                               const float* __restrict__ counts,
                               const float* __restrict__ W,   // [64][128] f32
                               float* __restrict__ out,
                               int row0, int N) {
    int t = blockIdx.x * blockDim.x + threadIdx.x;
    int row = row0 + (t >> 7);
    int col = t & (D_OUT - 1);
    if (row >= N) return;
    float inv = 1.0f / (counts[row] + 1.0f);
    float accf = 0.0f;
    for (int k = 0; k < D_IN; ++k) {
        float m = (summed[(size_t)row * D_IN + k] + x[(size_t)row * D_IN + k]) * inv;
        accf += m * W[k * D_OUT + col];
    }
    out[(size_t)row * D_OUT + col] = accf;
}

extern "C" void kernel_launch(void* const* d_in, const int* in_sizes, int n_in,
                              void* d_out, int out_size, void* d_ws, size_t ws_size,
                              hipStream_t stream) {
    const float* x  = (const float*)d_in[0];
    const int*   ei = (const int*)d_in[1];
    const float* W  = (const float*)d_in[2];
    float* out = (float*)d_out;

    int N = in_sizes[0] / D_IN;   // 50000
    int E = in_sizes[1] / 2;      // 800000

    // workspace layout: summed[N*64] f32 | counts[N] f32 | Wt[128*64] f16
    float* summed = (float*)d_ws;
    float* counts = summed + (size_t)N * D_IN;
    size_t wt_off = (((size_t)N * (D_IN + 1)) * sizeof(float) + 255) & ~(size_t)255;
    _Float16* Wt  = (_Float16*)((char*)d_ws + wt_off);

    int nz = N * (D_IN + 1);
    ma_zero_kernel<<<(nz + 255) / 256, 256, 0, stream>>>(summed, nz);
    ma_wt_kernel<<<(D_IN * D_OUT + 255) / 256, 256, 0, stream>>>(W, Wt);

    long long st = (long long)E * D_IN;
    ma_scatter_kernel<<<(unsigned)((st + 255) / 256), 256, 0, stream>>>(
        x, ei, summed, counts, E);

    int n_full = N / 16;          // 3125 full tiles (exact for N=50000)
    if (n_full > 0) {
        int threads = 128;        // 4 waves per block
        int blocks  = (n_full + 3) / 4;
        ma_gemm_wmma_kernel<<<blocks, threads, 0, stream>>>(
            x, summed, counts, Wt, out, n_full);
    }
    int rem = N - n_full * 16;    // 0 here; generic ragged tail
    if (rem > 0) {
        int tthreads = rem * D_OUT;
        ma_tail_kernel<<<(tthreads + 127) / 128, 128, 0, stream>>>(
            x, summed, counts, W, out, n_full * 16, N);
    }
}
